// MultiQueryAttention_16200616641137
// MI455X (gfx1250) — compile-verified, matrix operand order checked by reading, not run on hardware
//
#include <hip/hip_runtime.h>
#include <hip/hip_bf16.h>

// MQA: B=2 S=2048 D=1024 H=16 DH=64 CACHE=2048 MAX_KV=4096.
// Memory-bound (~85MB of mandatory HBM traffic -> ~4us floor at 23.3TB/s).
// GEMM-shaped work on v_wmma_f32_16x16x32_f16 (f32 accum). kh/vh (validated
// outputs) kept in exact fp32. Reference mask j<=i => attention touches ONLY
// the cache (j<2048). Attention blocks of 8 waves share K/V chunks staged in
// LDS via global_load_async_to_lds_b128 + s_wait_asynccnt (cuts K/V L2
// traffic ~8x vs per-wave streaming).

constexpr int Bb = 2, S = 2048, Dd = 1024, H = 16, DH = 64, CACHE = 2048, MAXKV = 4096;

typedef __attribute__((ext_vector_type(16))) _Float16 v16h;
typedef __attribute__((ext_vector_type(8)))  _Float16 h8;
typedef __attribute__((ext_vector_type(8)))  float    v8f;

// Load a 16x32 f16 A/B fragment slice for this lane. Per ISA layout:
// element e<8  -> K = 8*g + e          (contiguous 16B run)
// element e>=8 -> K = 16 + 8*g + (e-8) (contiguous 16B run)
__device__ __forceinline__ v16h load_frag16(const _Float16* base, int g) {
  h8 lo = *(const h8*)(base + 8 * g);
  h8 hi = *(const h8*)(base + 16 + 8 * g);
  v16h r;
#pragma unroll
  for (int i = 0; i < 8; ++i) { r[i] = lo[i]; r[i + 8] = hi[i]; }
  return r;
}

__device__ __forceinline__ v8f wmma32f16(v16h a, v16h b, v8f c) {
  // (neg_a, A, neg_b, B, c_mod, C, reuse_a, reuse_b)
  return __builtin_amdgcn_wmma_f32_16x16x32_f16(false, a, false, b, (short)0, c,
                                                false, false);
}

// Async 16B copy global -> LDS (no VGPR round trip, tracked by ASYNCcnt).
// Generic pointer to __shared__ carries the wave-relative LDS byte offset in
// its low 32 bits (flat->LDS mapping truncates to addr[31:0]).
__device__ __forceinline__ void async_copy16(unsigned int lds_addr, const void* gaddr) {
  asm volatile("global_load_async_to_lds_b128 %0, %1, off"
               :
               : "v"(lds_addr), "v"(gaddr)
               : "memory");
}

__device__ __forceinline__ void wait_async0() {
  asm volatile("s_wait_asynccnt 0x0" ::: "memory");
}

// ---------------- converts ----------------
__global__ void cvt_f32_f16(const float* __restrict__ s, _Float16* __restrict__ d, int n) {
  int i = blockIdx.x * blockDim.x + threadIdx.x;
  int stride = gridDim.x * blockDim.x;
  for (; i < n; i += stride) d[i] = (_Float16)s[i];
}

// cache_v[b][j][d] -> Vt[b][d][j] (f16), so V B-frags are contiguous 16B runs.
__global__ void cvt_v_transpose(const float* __restrict__ cv, _Float16* __restrict__ Vt) {
  const int n = Bb * CACHE * DH;
  int i = blockIdx.x * blockDim.x + threadIdx.x;
  int stride = gridDim.x * blockDim.x;
  for (; i < n; i += stride) {
    int b = i / (CACHE * DH);
    int r = i - b * (CACHE * DH);
    int j = r / DH, d = r - j * DH;
    Vt[(size_t)b * CACHE * DH + (size_t)d * CACHE + j] = (_Float16)cv[i];
  }
}

// kc[b][0..CACHE) = cache_k (contiguous prefix of each batch's region), same for vc.
__global__ void copy_cache(const float* __restrict__ ck, const float* __restrict__ cv,
                           float* __restrict__ kcO, float* __restrict__ vcO) {
  const int n = Bb * CACHE * DH;
  int i = blockIdx.x * blockDim.x + threadIdx.x;
  int stride = gridDim.x * blockDim.x;
  for (; i < n; i += stride) {
    int b = i / (CACHE * DH);
    int r = i - b * (CACHE * DH);
    size_t o = (size_t)b * MAXKV * DH + r;
    kcO[o] = ck[i];
    vcO[o] = cv[i];
  }
}

// ---------------- kh/vh projection (exact fp32; fills kc/vc tail) ----------------
__global__ void kv_proj(const float* __restrict__ k, const float* __restrict__ v,
                        const float* __restrict__ wk, const float* __restrict__ wv,
                        float* __restrict__ kcO, float* __restrict__ vcO) {
  int t = threadIdx.x;
  int d = t & 63, rs = t >> 6;          // 64 cols x 4 row-slots
  int base = blockIdx.x * 16;           // 16 rows / block
  for (int rr = 0; rr < 4; ++rr) {
    int row = base + rr * 4 + rs;       // row in [0, B*S)
    int b = row >> 11, s = row & 2047;
    const float* kr = k + (size_t)row * Dd;
    const float* vr = v + (size_t)row * Dd;
    float aK = 0.f, aV = 0.f;
#pragma unroll 8
    for (int kk = 0; kk < Dd; ++kk) {
      aK = fmaf(kr[kk], wk[kk * DH + d], aK);
      aV = fmaf(vr[kk], wv[kk * DH + d], aV);
    }
    size_t o = (size_t)b * MAXKV * DH + (size_t)(CACHE + s) * DH + d;
    kcO[o] = aK;
    vcO[o] = aV;
  }
}

// ---------------- WMMA GEMM: C(MxN) = A(MxK,f16) * B(KxN,f16) ----------------
// block = 256 (8 waves stacked in M). Per wave: 16x64 tile (4 f32 accum frags).
// B k-slab staged transposed in LDS so B-frags read as two ds_load_b128.
// mode 0: write f16 head-major Qh[b][head][s][d];  mode 1: write f32 row-major.
__global__ void gemm_wmma_f16(const _Float16* __restrict__ A, const _Float16* __restrict__ Bm,
                              void* __restrict__ outp, int M, int N, int K, int mode) {
  __shared__ __align__(16) _Float16 Bts[64 * 32];   // [col][k] transposed slab
  int t = threadIdx.x;
  int wave = t >> 5, lane = t & 31, g = lane >> 4, ln = lane & 15;
  int mtile = blockIdx.y * 8 + wave;
  int row0 = mtile * 16, n0 = blockIdx.x * 64;
  v8f acc[4] = {};
  const _Float16* arow = A + (size_t)(row0 + ln) * K;

  for (int k0 = 0; k0 < K; k0 += 32) {
    {  // cooperative transpose-stage: 256 thr x 8 halves = 32x64 slab
      int r = t >> 3, cb = (t & 7) * 8;
      h8 tmp = *(const h8*)(Bm + (size_t)(k0 + r) * N + n0 + cb);
#pragma unroll
      for (int i = 0; i < 8; ++i) Bts[(cb + i) * 32 + r] = tmp[i];
      if (k0 + 32 < K)
        __builtin_prefetch(Bm + (size_t)(k0 + 32 + r) * N + n0 + cb, 0, 1);
    }
    __syncthreads();
    v16h a = load_frag16(arow + k0, g);
#pragma unroll
    for (int t4 = 0; t4 < 4; ++t4) {
      v16h bf = load_frag16(&Bts[(t4 * 16 + ln) * 32], g);
      acc[t4] = wmma32f16(a, bf, acc[t4]);
    }
    __syncthreads();
  }

#pragma unroll
  for (int t4 = 0; t4 < 4; ++t4)
#pragma unroll
    for (int r = 0; r < 8; ++r) {
      int m = r + 8 * g;
      int R = row0 + m;
      if (mode == 0) {
        _Float16* O = (_Float16*)outp;
        int b = R >> 11, s = R & 2047;
        int head = blockIdx.x;            // N==1024 -> one head per 64-col tile
        int d = t4 * 16 + ln;
        O[(((size_t)(b * H + head)) * S + s) * DH + d] = (_Float16)acc[t4][r];
      } else {
        float* O = (float*)outp;
        O[(size_t)R * N + n0 + t4 * 16 + ln] = acc[t4][r];
      }
    }
}

// ---------------- flash attention over the cache ----------------
// Block = 8 waves = 128 consecutive queries of one (b,h). Each 32-kv chunk is
// staged once per block: K chunk (32 rows x 64 d) and V chunk (64 d x 32 kv,
// pre-transposed) via async load-to-LDS, then every wave runs 4 score WMMAs +
// 4 PV WMMAs against the shared tiles. Online softmax with width-16 shuffles.
__global__ void attn_mqa(const _Float16* __restrict__ Kf, const _Float16* __restrict__ Vt,
                         const _Float16* __restrict__ Qh, _Float16* __restrict__ valsF) {
  __shared__ __align__(16) _Float16 Ks[32 * 64];        // [kv][d]
  __shared__ __align__(16) _Float16 Vs[64 * 32];        // [d][kv]
  __shared__ __align__(16) _Float16 Plds[8][16 * 32];   // per-wave P relayout
  int t = threadIdx.x;
  int wv = t >> 5, lane = t & 31, g = lane >> 4, ln = lane & 15;
  int qblk = blockIdx.x & 15;            // 16 query-blocks per (b,h)
  int bh = blockIdx.x >> 4;
  int b = bh >> 4, h = bh & 15;
  int q0b = qblk * 128;
  int q0 = q0b + wv * 16;                // this wave's 16 queries
  _Float16* myP = Plds[wv];

  const _Float16* qrow = Qh + (((size_t)(b * H + h)) * S + q0 + ln) * DH;
  v16h aq0 = load_frag16(qrow, g);        // K-slice 0..31 of DH
  v16h aq1 = load_frag16(qrow + 32, g);   // K-slice 32..63

  // staging addresses: 256 threads x 16B cover each 4KB chunk
  int kr = t >> 3, kcb = (t & 7) * 8;     // K: kv-row 0..31, d-halves
  int vd = t >> 2, vkb = (t & 3) * 8;     // V: d-row 0..63, kv-halves
  unsigned int ldsK = (unsigned int)(uintptr_t)(Ks + kr * 64 + kcb);
  unsigned int ldsV = (unsigned int)(uintptr_t)(Vs + vd * 32 + vkb);
  const _Float16* gK0 = Kf + (size_t)b * CACHE * DH + (size_t)kr * DH + kcb;
  const _Float16* gV0 = Vt + (size_t)b * CACHE * DH + (size_t)vd * CACHE + vkb;

  float mr[8], lr[8];
#pragma unroll
  for (int r = 0; r < 8; ++r) { mr[r] = -__builtin_inff(); lr[r] = 0.f; }
  v8f acc[4] = {};

  const float scale = 0.125f;             // 1/sqrt(64)
  const int mylim = q0 + 16;              // causal: this wave needs kv < mylim
  const int blklim = q0b + 128;           // block-uniform loop bound (<= CACHE)
  for (int kv0 = 0; kv0 < blklim; kv0 += 32) {
    async_copy16(ldsK, gK0 + (size_t)kv0 * DH);   // K rows kv0..kv0+31
    async_copy16(ldsV, gV0 + kv0);                // V cols kv0..kv0+31
    wait_async0();
    __syncthreads();

    if (kv0 < mylim) {
      float sv[2][8];
#pragma unroll
      for (int c = 0; c < 2; ++c) {
        v8f s = {};
        const _Float16* krow = Ks + (c * 16 + ln) * 64;
        s = wmma32f16(aq0, load_frag16(krow, g), s);
        s = wmma32f16(aq1, load_frag16(krow + 32, g), s);
        int j = kv0 + c * 16 + ln;
#pragma unroll
        for (int r = 0; r < 8; ++r) {
          int i = q0 + r + 8 * g;
          sv[c][r] = (j <= i) ? s[r] * scale : -__builtin_inff();
        }
      }
      // online softmax (row stats replicated across each 16-lane group)
#pragma unroll
      for (int r = 0; r < 8; ++r) {
        float x = fmaxf(sv[0][r], sv[1][r]);
#pragma unroll
        for (int off = 1; off < 16; off <<= 1) x = fmaxf(x, __shfl_xor(x, off, 16));
        float mnew = fmaxf(mr[r], x);
        float esc = __expf(mr[r] - mnew);
        float p0 = __expf(sv[0][r] - mnew);
        float p1 = __expf(sv[1][r] - mnew);
        float rs = p0 + p1;
#pragma unroll
        for (int off = 1; off < 16; off <<= 1) rs += __shfl_xor(rs, off, 16);
        lr[r] = lr[r] * esc + rs;
        mr[r] = mnew;
        int m = r + 8 * g;
        myP[m * 32 + ln]      = (_Float16)p0;   // P tile 16x32, row m
        myP[m * 32 + 16 + ln] = (_Float16)p1;
#pragma unroll
        for (int t4 = 0; t4 < 4; ++t4) acc[t4][r] *= esc;
      }
      // P as A-fragment (same-wave LDS round trip; DS ops in-order per wave)
      v16h ap = load_frag16(myP + ln * 32, g);
#pragma unroll
      for (int t4 = 0; t4 < 4; ++t4)
        acc[t4] = wmma32f16(ap, load_frag16(Vs + (t4 * 16 + ln) * 32, g), acc[t4]);
    }
    __syncthreads();   // protect Ks/Vs before next chunk's staging
  }

#pragma unroll
  for (int t4 = 0; t4 < 4; ++t4)
#pragma unroll
    for (int r = 0; r < 8; ++r) {
      int m = r + 8 * g;
      float o = acc[t4][r] / lr[r];
      valsF[((size_t)(b * S + q0 + m)) * (H * DH) + h * DH + t4 * 16 + ln] = (_Float16)o;
    }
}

extern "C" void kernel_launch(void* const* d_in, const int* in_sizes, int n_in,
                              void* d_out, int out_size, void* d_ws, size_t ws_size,
                              hipStream_t stream) {
  (void)in_sizes; (void)n_in; (void)out_size; (void)ws_size;
  const float* q       = (const float*)d_in[0];
  const float* k       = (const float*)d_in[1];
  const float* v       = (const float*)d_in[2];
  const float* cache_k = (const float*)d_in[3];
  const float* cache_v = (const float*)d_in[4];
  const float* wq      = (const float*)d_in[5];
  const float* wk      = (const float*)d_in[6];
  const float* wv      = (const float*)d_in[7];
  const float* wo      = (const float*)d_in[8];

  float* out = (float*)d_out;                      // (B,S,D)
  float* kcO = out + (size_t)Bb * S * Dd;          // (B,1,MAXKV,DH)
  float* vcO = kcO + (size_t)Bb * MAXKV * DH;

  _Float16* ws    = (_Float16*)d_ws;               // ~30.4 MB total
  _Float16* qf    = ws;                            // 4,194,304
  _Float16* wqf   = qf  + 4194304;                 // 1,048,576
  _Float16* wof   = wqf + 1048576;                 // 1,048,576
  _Float16* Kf    = wof + 1048576;                 //   262,144
  _Float16* Vt    = Kf  + 262144;                  //   262,144
  _Float16* Qhf   = Vt  + 262144;                  // 4,194,304
  _Float16* valsF = Qhf + 4194304;                 // 4,194,304

  cvt_f32_f16<<<2048, 256, 0, stream>>>(q,  qf,  Bb * S * Dd);
  cvt_f32_f16<<<1024, 256, 0, stream>>>(wq, wqf, Dd * H * DH);
  cvt_f32_f16<<<1024, 256, 0, stream>>>(wo, wof, H * DH * Dd);
  cvt_f32_f16<<<256,  256, 0, stream>>>(cache_k, Kf, Bb * CACHE * DH);
  cvt_v_transpose<<<256, 256, 0, stream>>>(cache_v, Vt);
  copy_cache<<<512, 256, 0, stream>>>(cache_k, cache_v, kcO, vcO);
  kv_proj<<<(Bb * S) / 16, 256, 0, stream>>>(k, v, wk, wv, kcO, vcO);

  // qh = q @ wq  (4096x1024x1024), f16 head-major output
  gemm_wmma_f16<<<dim3(16, 32), 256, 0, stream>>>(qf, wqf, (void*)Qhf,
                                                  Bb * S, H * DH, Dd, 0);
  // flash attention: 512 blocks x 8 waves (128 queries per block)
  attn_mqa<<<512, 256, 0, stream>>>(Kf, Vt, Qhf, valsF);
  // out = vals @ wo (4096x1024x1024), f32 output
  gemm_wmma_f16<<<dim3(16, 32), 256, 0, stream>>>(valsF, wof, (void*)out,
                                                  Bb * S, Dd, H * DH, 1);
}
